// SpatioTemporalGNN_24696061952235
// MI455X (gfx1250) — matrix operand order repaired and checked by reading, not compiled, not run on hardware
//
#include <hip/hip_runtime.h>
#include <hip/hip_bf16.h>

#define NN 50000
#define EE 800000
#define TT 12
#define HH 128
#define NHEADS 4
#define CCH 32
#define LLAY 3
#define OO 12
#define G3 (3 * HH)

typedef __attribute__((ext_vector_type(16))) _Float16 v16h;
typedef __attribute__((ext_vector_type(8)))  _Float16 v8h;
typedef __attribute__((ext_vector_type(8)))  float    v8f;
typedef __attribute__((ext_vector_type(4)))  float    v4f;

// ---------------------------------------------------------------------------
// WMMA fragment helpers (CDNA5 wave32, V_WMMA_F32_16X16X32_F16)
//
// A 16x32 f16 layout: lane l holds row m = l&15; halves e=0..7 -> k = kc*32 +
// 8*(l>>4) + e ; halves e=8..15 -> k += 16.  Two contiguous 16B loads.
// B 32x16 f16 layout: lane l holds col n = l&15; same k order as A:
// e=0..7 -> k = kc*32 + 8*(l>>4) + e; e=8..15 -> k += 16.  Two 16B loads (weights stored [n][k]).
// C/D 16x16 f32: vgpr i, lane l -> (m = i + 8*(l>>4), n = l&15).
// ---------------------------------------------------------------------------
__device__ __forceinline__ v16h frag_from2(v8h lo, v8h hi) {
  v16h r;
#pragma unroll
  for (int i = 0; i < 8; ++i) { r[i] = lo[i]; r[i + 8] = hi[i]; }
  return r;
}

__device__ __forceinline__ v16h load_a_frag(const _Float16* base, int kc, int lane) {
  const int m = lane & 15, hi = lane >> 4;
  const _Float16* p = base + (size_t)m * HH + kc * 32 + hi * 8;
  return frag_from2(*(const v8h*)p, *(const v8h*)(p + 16));
}

__device__ __forceinline__ v16h load_b_frag(const _Float16* wrow, int kc, int lane) {
  const int n = lane & 15, hi = lane >> 4;
  const _Float16* p = wrow + (size_t)n * HH + kc * 32 + hi * 8;
  return frag_from2(*(const v8h*)p, *(const v8h*)(p + 16));
}

__device__ __forceinline__ v8f wmma_f16(v16h a, v16h b, v8f c) {
  return __builtin_amdgcn_wmma_f32_16x16x32_f16(false, a, false, b, (short)0, c, false, false);
}

// Fast transcendentals: v_exp_f32 + v_rcp_f32, no IEEE divide fixup chains.
__device__ __forceinline__ float rcp_(float x) { return __builtin_amdgcn_rcpf(x); }
__device__ __forceinline__ float sigmoid_(float v) { return rcp_(1.f + __expf(-v)); }
__device__ __forceinline__ float tanh_(float v) { return 1.f - 2.f * rcp_(1.f + __expf(2.f * v)); }

// ---------------------------------------------------------------------------
// Weight prep: f32 -> f16 copy, and f32 [K][n] -> f16 [nPad][K] transpose.
// ---------------------------------------------------------------------------
__global__ void cvt16_kernel(const float* __restrict__ in, _Float16* __restrict__ out, int n) {
  int i = blockIdx.x * blockDim.x + threadIdx.x;
  if (i < n) out[i] = (_Float16)in[i];
}

__global__ void transpose16_kernel(const float* __restrict__ in, _Float16* __restrict__ out,
                                   int rows, int cols, int colsPad) {
  int idx = blockIdx.x * blockDim.x + threadIdx.x;
  if (idx >= colsPad * rows) return;
  int j = idx / rows, i = idx - j * rows;                 // j = out col index, i = k
  out[(size_t)j * rows + i] = (j < cols) ? (_Float16)in[(size_t)i * cols + j] : (_Float16)0.f;
}

// ---------------------------------------------------------------------------
// GRU layer 0: input dim 1 (gx is an outer product), recurrence via WMMA.
// Block = 128 threads (4 waves), each wave owns one 16-node tile.
// Dyn LDS: Whh f16 [384][128] + Wih0 f32 [384] + 16KB per wave (hA,hB,r,z).
// ---------------------------------------------------------------------------
__global__ void gru0_kernel(const float* __restrict__ x, const float* __restrict__ wih,
                            const float* __restrict__ bih, const float* __restrict__ bhh,
                            const _Float16* __restrict__ whh16, _Float16* __restrict__ hs0) {
  extern __shared__ char smem[];
  _Float16* Wh = (_Float16*)smem;                       // 384*128 f16
  float* Wi = (float*)(smem + G3 * HH * 2);             // 384 f32
  char* pw = smem + G3 * HH * 2 + G3 * 4;
  const int lane = threadIdx.x & 31;
  const int wv = threadIdx.x >> 5;
  _Float16* hA = (_Float16*)(pw + wv * 16384);
  _Float16* hB = hA + 2048;
  _Float16* rB = hA + 4096;
  _Float16* zB = hA + 6144;

  for (int i = threadIdx.x; i < G3 * HH / 8; i += blockDim.x)
    ((v8h*)Wh)[i] = ((const v8h*)whh16)[i];
  for (int i = threadIdx.x; i < G3; i += blockDim.x) Wi[i] = wih[i];
  __syncthreads();

  const int node0 = (blockIdx.x * 4 + wv) * 16;
  if (node0 >= NN) return;                              // wave-uniform

  for (int i = lane; i < 2048; i += 32) hA[i] = (_Float16)0.f;

  const int n = lane & 15;
  const int mh = (lane >> 4) * 8;

  for (int t = 0; t < TT; ++t) {
    _Float16* hc = (t & 1) ? hB : hA;
    _Float16* hn = (t & 1) ? hA : hB;

    // Hoist per-timestep invariants: A fragments of h, and this step's x row.
    v16h ah[4];
#pragma unroll
    for (int kc = 0; kc < 4; ++kc) ah[kc] = load_a_frag(hc, kc, lane);
    float xv[8];
#pragma unroll
    for (int i = 0; i < 8; ++i) xv[i] = x[(size_t)(node0 + i + mh) * TT + t];

    for (int chunk = 0; chunk < 2; ++chunk) {           // 0 = r, 1 = z
      _Float16* dstb = chunk ? zB : rB;
      for (int j = 0; j < 8; ++j) {
        v8f acc = {};
#pragma unroll
        for (int kc = 0; kc < 4; ++kc)
          acc = wmma_f16(ah[kc],
                         load_b_frag(Wh + (size_t)(chunk * HH + j * 16) * HH, kc, lane), acc);
        const int g = chunk * HH + j * 16 + n;
        const float bsum = bih[g] + bhh[g];
        const float wi = Wi[g];
#pragma unroll
        for (int i = 0; i < 8; ++i)
          dstb[(i + mh) * HH + j * 16 + n] = (_Float16)sigmoid_(xv[i] * wi + bsum + acc[i]);
      }
    }
    for (int j = 0; j < 8; ++j) {                        // n gate
      v8f acc = {};
#pragma unroll
      for (int kc = 0; kc < 4; ++kc)
        acc = wmma_f16(ah[kc],
                       load_b_frag(Wh + (size_t)(2 * HH + j * 16) * HH, kc, lane), acc);
      const int g = 2 * HH + j * 16 + n;
      const float bi = bih[g], bh = bhh[g], wi = Wi[g];
#pragma unroll
      for (int i = 0; i < 8; ++i) {
        const int m = i + mh;
        const int col = j * 16 + n;
        const float gx = xv[i] * wi + bi;
        const float gh = acc[i] + bh;
        const float rv = (float)rB[m * HH + col];
        const float nv = tanh_(gx + rv * gh);
        const float zv = (float)zB[m * HH + col];
        const float ho = (float)hc[m * HH + col];
        const float hv = (1.f - zv) * nv + zv * ho;
        hn[m * HH + col] = (_Float16)hv;
        hs0[((size_t)t * NN + node0 + m) * HH + col] = (_Float16)hv;
      }
    }
  }
}

// ---------------------------------------------------------------------------
// GRU layer 1: fused input GEMM (A from global hs0[t]) + recurrent GEMM.
// Dyn LDS: Wih + Whh f16 (192KB) + 16KB per wave.
// ---------------------------------------------------------------------------
__global__ void gru1_kernel(const _Float16* __restrict__ hs0, const _Float16* __restrict__ wih16,
                            const _Float16* __restrict__ whh16, const float* __restrict__ bih,
                            const float* __restrict__ bhh, float* __restrict__ h32,
                            _Float16* __restrict__ h16) {
  extern __shared__ char smem[];
  _Float16* Wi = (_Float16*)smem;                       // [384][128]
  _Float16* Wh = Wi + (size_t)G3 * HH;                  // [384][128]
  char* pw = smem + 2 * G3 * HH * 2;
  const int lane = threadIdx.x & 31;
  const int wv = threadIdx.x >> 5;
  _Float16* hA = (_Float16*)(pw + wv * 16384);
  _Float16* hB = hA + 2048;
  _Float16* rB = hA + 4096;
  _Float16* zB = hA + 6144;

  for (int i = threadIdx.x; i < G3 * HH / 8; i += blockDim.x) {
    ((v8h*)Wi)[i] = ((const v8h*)wih16)[i];
    ((v8h*)Wh)[i] = ((const v8h*)whh16)[i];
  }
  __syncthreads();

  const int node0 = (blockIdx.x * 4 + wv) * 16;
  if (node0 >= NN) return;

  for (int i = lane; i < 2048; i += 32) hA[i] = (_Float16)0.f;

  const int n = lane & 15;
  const int mh = (lane >> 4) * 8;

  for (int t = 0; t < TT; ++t) {
    _Float16* hc = (t & 1) ? hB : hA;
    _Float16* hn = (t & 1) ? hA : hB;
    const _Float16* xg = hs0 + ((size_t)t * NN + node0) * HH;

    // Hoist A fragments of both GEMM inputs for this timestep.
    v16h ahx[4], ahh[4];
#pragma unroll
    for (int kc = 0; kc < 4; ++kc) {
      ahx[kc] = load_a_frag(xg, kc, lane);
      ahh[kc] = load_a_frag(hc, kc, lane);
    }

    for (int chunk = 0; chunk < 2; ++chunk) {           // r, z: merge input+recurrent
      _Float16* dstb = chunk ? zB : rB;
      for (int j = 0; j < 8; ++j) {
        v8f acc = {};
#pragma unroll
        for (int kc = 0; kc < 4; ++kc)
          acc = wmma_f16(ahx[kc],
                         load_b_frag(Wi + (size_t)(chunk * HH + j * 16) * HH, kc, lane), acc);
#pragma unroll
        for (int kc = 0; kc < 4; ++kc)
          acc = wmma_f16(ahh[kc],
                         load_b_frag(Wh + (size_t)(chunk * HH + j * 16) * HH, kc, lane), acc);
        const int g = chunk * HH + j * 16 + n;
        const float bsum = bih[g] + bhh[g];
#pragma unroll
        for (int i = 0; i < 8; ++i)
          dstb[(i + mh) * HH + j * 16 + n] = (_Float16)sigmoid_(acc[i] + bsum);
      }
    }
    for (int j = 0; j < 8; ++j) {                        // n gate: keep separate
      v8f ax = {}, ah = {};
#pragma unroll
      for (int kc = 0; kc < 4; ++kc)
        ax = wmma_f16(ahx[kc],
                      load_b_frag(Wi + (size_t)(2 * HH + j * 16) * HH, kc, lane), ax);
#pragma unroll
      for (int kc = 0; kc < 4; ++kc)
        ah = wmma_f16(ahh[kc],
                      load_b_frag(Wh + (size_t)(2 * HH + j * 16) * HH, kc, lane), ah);
      const int g = 2 * HH + j * 16 + n;
      const float bi = bih[g], bh = bhh[g];
#pragma unroll
      for (int i = 0; i < 8; ++i) {
        const int m = i + mh;
        const int col = j * 16 + n;
        const float gx = ax[i] + bi;
        const float gh = ah[i] + bh;
        const float rv = (float)rB[m * HH + col];
        const float nv = tanh_(gx + rv * gh);
        const float zv = (float)zB[m * HH + col];
        const float ho = (float)hc[m * HH + col];
        const float hv = (1.f - zv) * nv + zv * ho;
        hn[m * HH + col] = (_Float16)hv;
        if (t == TT - 1) {
          h32[(size_t)(node0 + m) * HH + col] = hv;
          h16[(size_t)(node0 + m) * HH + col] = (_Float16)hv;
        }
      }
    }
  }
}

// ---------------------------------------------------------------------------
// Generic 16x16-tile GEMM: out[row, col] = A(f16 [N][128]) x Wt(f16 [nPad][128])
// + bias, optional relu, f32 and/or f16 outputs.  One wave per block.
// ---------------------------------------------------------------------------
__global__ void gemm128_kernel(const _Float16* __restrict__ A, const _Float16* __restrict__ Wt,
                               const float* __restrict__ bias, float* __restrict__ out32,
                               int ldo, _Float16* __restrict__ out16, int nOut, int relu) {
  const int lane = threadIdx.x & 31;
  const int row0 = blockIdx.x * 16;
  const int col0 = blockIdx.y * 16;
  v8f acc = {};
  const _Float16* abase = A + (size_t)row0 * HH;
#pragma unroll
  for (int kc = 0; kc < 4; ++kc)
    acc = wmma_f16(load_a_frag(abase, kc, lane),
                   load_b_frag(Wt + (size_t)col0 * HH, kc, lane), acc);
  const int nl = lane & 15;
  const int col = col0 + nl;
  const bool ok = col < nOut;
  const float bv = ok ? bias[col] : 0.f;
#pragma unroll
  for (int i = 0; i < 8; ++i) {
    const int row = row0 + i + ((lane >> 4) * 8);
    float v = acc[i] + bv;
    if (relu) v = fmaxf(v, 0.f);
    if (ok) {
      if (out32) out32[(size_t)row * ldo + col] = v;
      if (out16) out16[(size_t)row * HH + col] = (_Float16)v;
    }
  }
}

// ---------------------------------------------------------------------------
// GAT edge phase (bandwidth-bound: vectorized 16B gathers + prefetch).
// ---------------------------------------------------------------------------
__global__ void zero_kernel(float* __restrict__ oacc, unsigned* __restrict__ smax,
                            float* __restrict__ denom) {
  int i = blockIdx.x * blockDim.x + threadIdx.x;
  if (i < NN * HH) oacc[i] = 0.f;
  if (i < NN * NHEADS) { smax[i] = 0u; denom[i] = 0.f; }
}

__device__ __forceinline__ unsigned fenc(float f) {
  unsigned b = __float_as_uint(f);
  return (b & 0x80000000u) ? ~b : (b | 0x80000000u);
}
__device__ __forceinline__ float fdec(unsigned u) {
  unsigned b = (u & 0x80000000u) ? (u ^ 0x80000000u) : ~u;
  return __uint_as_float(b);
}

__global__ void edge_score_kernel(const int* __restrict__ ei, const float* __restrict__ ea,
                                  const float* __restrict__ xl, const float* __restrict__ xr,
                                  const float* __restrict__ We, const float* __restrict__ att,
                                  float* __restrict__ s, unsigned* __restrict__ smax) {
  int e = blockIdx.x * blockDim.x + threadIdx.x;
  if (e >= EE) return;
  const int src = ei[e], dst = ei[EE + e];
  const float a = ea[e];
  const v4f* pl = (const v4f*)(xl + (size_t)src * HH);
  const v4f* pr = (const v4f*)(xr + (size_t)dst * HH);
  const v4f* pe = (const v4f*)We;
  const v4f* pa = (const v4f*)att;
  __builtin_prefetch(pl, 0, 0);                          // global_prefetch_b8
  __builtin_prefetch(pr, 0, 0);
#pragma unroll
  for (int h = 0; h < NHEADS; ++h) {
    float sum = 0.f;
#pragma unroll
    for (int q = 0; q < CCH / 4; ++q) {
      const int k = h * (CCH / 4) + q;
      const v4f ml = pl[k], mr = pr[k], we = pe[k], at = pa[k];
#pragma unroll
      for (int u = 0; u < 4; ++u) {
        float m = ml[u] + mr[u] + a * we[u];
        m = (m > 0.f) ? m : 0.2f * m;                    // leaky_relu(0.2)
        sum += m * at[u];
      }
    }
    s[(size_t)e * NHEADS + h] = sum;
    atomicMax(smax + (size_t)dst * NHEADS + h, fenc(sum));
  }
}

__global__ void edge_exp_kernel(const int* __restrict__ ei, float* __restrict__ s,
                                const unsigned* __restrict__ smax, float* __restrict__ denom) {
  int idx = blockIdx.x * blockDim.x + threadIdx.x;
  if (idx >= EE * NHEADS) return;
  const int e = idx >> 2, h = idx & 3;
  const int dst = ei[EE + e];
  const float es = __expf(s[idx] - fdec(smax[(size_t)dst * NHEADS + h]));
  s[idx] = es;                                           // overwrite in place
  atomicAdd(denom + (size_t)dst * NHEADS + h, es);
}

__global__ void edge_aggr_kernel(const int* __restrict__ ei, const float* __restrict__ es,
                                 const float* __restrict__ denom, const float* __restrict__ xl,
                                 float* __restrict__ oacc) {
  int idx = blockIdx.x * blockDim.x + threadIdx.x;
  if (idx >= EE * NHEADS) return;
  const int e = idx >> 2, h = idx & 3;
  const int src = ei[e], dst = ei[EE + e];
  const float d = fmaxf(denom[(size_t)dst * NHEADS + h], 1e-16f);
  const float alpha = es[idx] * rcp_(d);
  const v4f* pl = (const v4f*)(xl + (size_t)src * HH + h * CCH);
  float* po = oacc + (size_t)dst * HH + h * CCH;
  __builtin_prefetch(pl, 0, 0);
#pragma unroll
  for (int q = 0; q < CCH / 4; ++q) {
    const v4f v = pl[q];
#pragma unroll
    for (int u = 0; u < 4; ++u) atomicAdd(po + q * 4 + u, alpha * v[u]);
  }
}

__global__ void node_update_kernel(const float* __restrict__ oacc, const float* __restrict__ gbias,
                                   const float* __restrict__ gamma, const float* __restrict__ beta,
                                   float* __restrict__ h32, _Float16* __restrict__ h16) {
  const int node = blockIdx.x;
  const int c = threadIdx.x;
  __shared__ float sd[HH];
  const float z = oacc[(size_t)node * HH + c] + gbias[c];
  sd[c] = z;
  __syncthreads();
  for (int st = 64; st > 0; st >>= 1) { if (c < st) sd[c] += sd[c + st]; __syncthreads(); }
  const float mu = sd[0] * (1.f / HH);
  __syncthreads();
  const float d = z - mu;
  sd[c] = d * d;
  __syncthreads();
  for (int st = 64; st > 0; st >>= 1) { if (c < st) sd[c] += sd[c + st]; __syncthreads(); }
  const float var = sd[0] * (1.f / HH);
  const float y = d * rsqrtf(var + 1e-5f) * gamma[c] + beta[c];
  const float el = (y > 0.f) ? y : (__expf(y) - 1.f);    // elu
  const float hv = el + h32[(size_t)node * HH + c];      // residual
  h32[(size_t)node * HH + c] = hv;
  h16[(size_t)node * HH + c] = (_Float16)hv;
}

// ---------------------------------------------------------------------------
extern "C" void kernel_launch(void* const* d_in, const int* in_sizes, int n_in,
                              void* d_out, int out_size, void* d_ws, size_t ws_size,
                              hipStream_t stream) {
  const float* x     = (const float*)d_in[0];
  const int*   ei    = (const int*)d_in[1];
  const float* ea    = (const float*)d_in[2];
  const float* wih0  = (const float*)d_in[3];
  const float* whh0  = (const float*)d_in[4];
  const float* bih0  = (const float*)d_in[5];
  const float* bhh0  = (const float*)d_in[6];
  const float* wih1  = (const float*)d_in[7];
  const float* whh1  = (const float*)d_in[8];
  const float* bih1  = (const float*)d_in[9];
  const float* bhh1  = (const float*)d_in[10];
  const float* gWl   = (const float*)d_in[11];
  const float* gbl   = (const float*)d_in[12];
  const float* gWr   = (const float*)d_in[13];
  const float* gbr   = (const float*)d_in[14];
  const float* gWe   = (const float*)d_in[15];
  const float* gatt  = (const float*)d_in[16];
  const float* gbias = (const float*)d_in[17];
  const float* lng   = (const float*)d_in[18];
  const float* lnb   = (const float*)d_in[19];
  const float* fW1   = (const float*)d_in[20];
  const float* fb1   = (const float*)d_in[21];
  const float* fW2   = (const float*)d_in[22];
  const float* fb2   = (const float*)d_in[23];
  float* out = (float*)d_out;

  char* ws = (char*)d_ws;
  size_t off = 0;
  auto alloc = [&](size_t bytes) -> void* {
    off = (off + 255) & ~(size_t)255;
    void* p = ws + off;
    off += bytes;
    return p;
  };
  _Float16* hs0   = (_Float16*)alloc((size_t)TT * NN * HH * 2);   // GRU0 sequence
  float*    h32   = (float*)alloc((size_t)NN * HH * 4);
  _Float16* h16   = (_Float16*)alloc((size_t)NN * HH * 2);
  float*    xl    = (float*)alloc((size_t)NN * HH * 4);
  float*    xr    = (float*)alloc((size_t)NN * HH * 4);
  float*    oacc  = (float*)alloc((size_t)NN * HH * 4);
  float*    sbuf  = (float*)alloc((size_t)EE * NHEADS * 4);
  unsigned* smax  = (unsigned*)alloc((size_t)NN * NHEADS * 4);
  float*    denom = (float*)alloc((size_t)NN * NHEADS * 4);
  _Float16* h1f16 = (_Float16*)alloc((size_t)NN * HH * 2);
  _Float16* whh0h = (_Float16*)alloc((size_t)G3 * HH * 2);
  _Float16* wih1h = (_Float16*)alloc((size_t)G3 * HH * 2);
  _Float16* whh1h = (_Float16*)alloc((size_t)G3 * HH * 2);
  _Float16* wlt   = (_Float16*)alloc((size_t)LLAY * HH * HH * 2);
  _Float16* wrt   = (_Float16*)alloc((size_t)LLAY * HH * HH * 2);
  _Float16* w1t   = (_Float16*)alloc((size_t)HH * HH * 2);
  _Float16* w2t   = (_Float16*)alloc((size_t)16 * HH * 2);

  // --- weight prep ---
  const int nc = G3 * HH;
  cvt16_kernel<<<(nc + 255) / 256, 256, 0, stream>>>(whh0, whh0h, nc);
  cvt16_kernel<<<(nc + 255) / 256, 256, 0, stream>>>(wih1, wih1h, nc);
  cvt16_kernel<<<(nc + 255) / 256, 256, 0, stream>>>(whh1, whh1h, nc);
  for (int l = 0; l < LLAY; ++l) {
    transpose16_kernel<<<(HH * HH + 255) / 256, 256, 0, stream>>>(
        gWl + (size_t)l * HH * HH, wlt + (size_t)l * HH * HH, HH, HH, HH);
    transpose16_kernel<<<(HH * HH + 255) / 256, 256, 0, stream>>>(
        gWr + (size_t)l * HH * HH, wrt + (size_t)l * HH * HH, HH, HH, HH);
  }
  transpose16_kernel<<<(HH * HH + 255) / 256, 256, 0, stream>>>(fW1, w1t, HH, HH, HH);
  transpose16_kernel<<<(16 * HH + 255) / 256, 256, 0, stream>>>(fW2, w2t, HH, OO, 16);

  // --- temporal GRU (WMMA) ---
  const int nodeTiles = NN / 16;                  // 3125
  const int gruBlocks = (nodeTiles + 3) / 4;      // 782 (4 waves/block)
  const size_t lds0 = (size_t)G3 * HH * 2 + G3 * 4 + 4 * 16384;   // ~162KB
  const size_t lds1 = (size_t)2 * G3 * HH * 2 + 4 * 16384;        // 256KB (<320KB/WGP)
  gru0_kernel<<<gruBlocks, 128, lds0, stream>>>(x, wih0, bih0, bhh0, whh0h, hs0);
  gru1_kernel<<<gruBlocks, 128, lds1, stream>>>(hs0, wih1h, whh1h, bih1, bhh1, h32, h16);

  // --- GATv2 layers ---
  for (int l = 0; l < LLAY; ++l) {
    gemm128_kernel<<<dim3(nodeTiles, HH / 16), 32, 0, stream>>>(
        h16, wlt + (size_t)l * HH * HH, gbl + (size_t)l * HH, xl, HH, (_Float16*)nullptr, HH, 0);
    gemm128_kernel<<<dim3(nodeTiles, HH / 16), 32, 0, stream>>>(
        h16, wrt + (size_t)l * HH * HH, gbr + (size_t)l * HH, xr, HH, (_Float16*)nullptr, HH, 0);
    zero_kernel<<<(NN * HH + 255) / 256, 256, 0, stream>>>(oacc, smax, denom);
    edge_score_kernel<<<EE / 256, 256, 0, stream>>>(
        ei, ea, xl, xr, gWe + (size_t)l * HH, gatt + (size_t)l * NHEADS * CCH, sbuf, smax);
    edge_exp_kernel<<<(EE * NHEADS) / 256, 256, 0, stream>>>(ei, sbuf, smax, denom);
    edge_aggr_kernel<<<(EE * NHEADS) / 256, 256, 0, stream>>>(ei, sbuf, denom, xl, oacc);
    node_update_kernel<<<NN, HH, 0, stream>>>(
        oacc, gbias + (size_t)l * HH, lng + (size_t)l * HH, lnb + (size_t)l * HH, h32, h16);
  }

  // --- MLP head (WMMA) ---
  gemm128_kernel<<<dim3(nodeTiles, HH / 16), 32, 0, stream>>>(
      h16, w1t, fb1, (float*)nullptr, 0, h1f16, HH, 1);
  gemm128_kernel<<<dim3(nodeTiles, 1), 32, 0, stream>>>(
      h1f16, w2t, fb2, out, OO, (_Float16*)nullptr, OO, 0);
}
